// TransformerBlock_10084583211545
// MI455X (gfx1250) — matrix operand order repaired and checked by reading, not compiled, not run on hardware
//
#include <hip/hip_runtime.h>
#include <hip/hip_bf16.h>
#include <math.h>

typedef __attribute__((ext_vector_type(16))) __bf16 v16bf;
typedef __attribute__((ext_vector_type(8)))  __bf16 v8bf;
typedef __attribute__((ext_vector_type(8)))  float  v8f;
typedef __attribute__((ext_vector_type(4)))  int    v4i;

#define DEVI __device__ __forceinline__

static constexpr int kB  = 2;
static constexpr int kS  = 2048;
static constexpr int kD  = 1024;
static constexpr int kH  = 16;
static constexpr int kF  = 4096;
static constexpr int kDK = 64;
static constexpr int kM  = kB * kS;   // 4096 rows
static constexpr int kBH = kB * kH;   // 32

// --- gfx1250 async load-to-LDS path (ASYNCcnt), guarded so compile is safe ---
#if defined(__has_builtin)
#  if __has_builtin(__builtin_amdgcn_global_load_async_to_lds_b128)
#    define HAVE_ASYNC 1
#  endif
#endif
#ifndef HAVE_ASYNC
#  define HAVE_ASYNC 0
#endif

DEVI void wait_async0() {
#if __has_builtin(__builtin_amdgcn_s_wait_asynccnt)
  __builtin_amdgcn_s_wait_asynccnt(0);
#else
  asm volatile("s_wait_asynccnt 0" ::: "memory");
#endif
}

// Copy 16 bytes global -> LDS (async if available, else load+ds_store).
DEVI void copy_b128(const __bf16* g, __bf16* l) {
#if HAVE_ASYNC
  __builtin_amdgcn_global_load_async_to_lds_b128(
      (v4i*)g, (__attribute__((address_space(3))) v4i*)l, 0, 0);
#else
  *(v8bf*)l = *(const v8bf*)g;
#endif
}

// ---------------------------------------------------------------------------
// WMMA fragment helpers (CDNA5 ISA §7.12.2 layouts, wave32)
// A (16x32 MxK, 16-bit): lane l (m=l&15, h=l>>4) holds K = {8h..8h+7} in
// VGPR0-3 and K = {16+8h..16+8h+7} in VGPR4-7  -> two 16B row chunks.
// B (32x16 KxN, 16-bit): lane l (n=l&15, h=l>>4) holds K = {8h..8h+7} and
// K = {16+8h..16+8h+7} of column n -> two 16B chunks (same order as A).
// C/D (16x16 f32): VGPR r, lane l -> m = r + 8*(l>>4), n = l&15.
// ---------------------------------------------------------------------------
DEVI v16bf load_a_frag(const __bf16* p, int ld) {
  int l = threadIdx.x & 31;
  int m = l & 15, h = l >> 4;
  v8bf lo = *(const v8bf*)(p + (size_t)m * ld + h * 8);
  v8bf hi = *(const v8bf*)(p + (size_t)m * ld + 16 + h * 8);
  v16bf r;
#pragma unroll
  for (int i = 0; i < 8; ++i) { r[i] = lo[i]; r[8 + i] = hi[i]; }
  return r;
}

DEVI v16bf load_b_frag(const __bf16* p, int ld) {
  int l = threadIdx.x & 31;
  int n = l & 15, h = l >> 4;
  v8bf lo = *(const v8bf*)(p + (size_t)n * ld + h * 8);
  v8bf hi = *(const v8bf*)(p + (size_t)n * ld + 16 + h * 8);
  v16bf r;
#pragma unroll
  for (int i = 0; i < 8; ++i) { r[i] = lo[i]; r[8 + i] = hi[i]; }
  return r;
}

DEVI v8f wmma_bf16(v16bf a, v16bf b, v8f c) {
  return __builtin_amdgcn_wmma_f32_16x16x32_bf16(
      /*neg_a=*/false, a, /*neg_b=*/false, b,
      /*c_mod=*/(short)0, c, /*reuse_a=*/false, /*reuse_b=*/false);
}

DEVI float gelu_exact(float x) {
  return 0.5f * x * (1.0f + erff(x * 0.70710678118654752f));
}

// ---------------------------------------------------------------------------
// fp32 -> bf16 conversion (weights)
// ---------------------------------------------------------------------------
__global__ void cvt_bf16_kernel(const float* __restrict__ in,
                                __bf16* __restrict__ out, int n) {
  int i = blockIdx.x * blockDim.x + threadIdx.x;
  int stride = gridDim.x * blockDim.x;
  for (; i < n; i += stride) out[i] = (__bf16)in[i];
}

// ---------------------------------------------------------------------------
// RMSNorm: one block per row of D=1024, 256 threads x float4
// ---------------------------------------------------------------------------
__global__ __launch_bounds__(256)
void rmsnorm_bf16_kernel(const float* __restrict__ x, const float* __restrict__ g,
                         __bf16* __restrict__ out) {
  int row = blockIdx.x;
  float4 v = ((const float4*)(x + (size_t)row * kD))[threadIdx.x];
  float ss = v.x * v.x + v.y * v.y + v.z * v.z + v.w * v.w;
  __shared__ float red[256];
  red[threadIdx.x] = ss;
  __syncthreads();
  for (int s = 128; s > 0; s >>= 1) {
    if (threadIdx.x < s) red[threadIdx.x] += red[threadIdx.x + s];
    __syncthreads();
  }
  float inv = rsqrtf(red[0] * (1.0f / kD) + 1e-5f);
  float4 gv = ((const float4*)g)[threadIdx.x];
  __bf16* o = out + (size_t)row * kD + threadIdx.x * 4;
  o[0] = (__bf16)(v.x * inv * gv.x);
  o[1] = (__bf16)(v.y * inv * gv.y);
  o[2] = (__bf16)(v.z * inv * gv.z);
  o[3] = (__bf16)(v.w * inv * gv.w);
}

// ---------------------------------------------------------------------------
// GEMM: C[M,N] = A[M,K] (bf16, row-major) * W[N,K]^T (bf16, row-major)
// Block = 256 thr = 8 waves; block tile 128x128; wave tile 64x32 (4x2 WMMAs).
// K-chunks of 32 staged cooperatively into LDS (128x32 A + 128x32 W tiles,
// double buffered, async load-to-LDS when available).
// STORE: 0 row-major bf16 | 1 Q/K heads [BH,S,64] | 2 V transposed [BH,64,S]
// EPI:   0 bf16 store     | 1 fp32 resid add      | 2 GELU -> bf16
// ---------------------------------------------------------------------------
// Stage one 128x32 bf16 tile: 256 threads x 2 x b128 (32B per thread).
DEVI void stage_tile(const __bf16* gbase, int K, __bf16* lbuf) {
  int t = threadIdx.x;
  int row = t >> 1, ch = t & 1;  // ch selects 16-element half of the row
  const __bf16* g = gbase + (size_t)row * K + ch * 16;
  __bf16* l = lbuf + row * 32 + ch * 16;
  copy_b128(g, l);
  copy_b128(g + 8, l + 8);
}

template <int STORE, int EPI>
__global__ __launch_bounds__(256)
void gemm_bf16_kernel(const __bf16* __restrict__ A, const __bf16* __restrict__ W,
                      int M, int N, int K,
                      float* __restrict__ outf, __bf16* __restrict__ outb,
                      const float* __restrict__ resid) {
  __shared__ __align__(64) __bf16 As[2][128 * 32];
  __shared__ __align__(64) __bf16 Bs[2][128 * 32];

  int wave = threadIdx.x >> 5;
  int lane = threadIdx.x & 31;
  int wm = wave & 1, wn = wave >> 1;
  int mbase = blockIdx.y * 128 + wm * 64;
  int nbase = blockIdx.x * 128 + wn * 32;
  const __bf16* Ab = A + (size_t)blockIdx.y * 128 * K;
  const __bf16* Wb = W + (size_t)blockIdx.x * 128 * K;

  v8f acc[4][2] = {};
  int nk = K >> 5;

  stage_tile(Ab, K, As[0]);
  stage_tile(Wb, K, Bs[0]);

  for (int kt = 0; kt < nk; ++kt) {
    int cur = kt & 1;
#if HAVE_ASYNC
    wait_async0();  // this wave's async copies into buf[cur] are done
#endif
    __syncthreads();  // everyone's copies are done, buf[cur] is valid
    if (kt + 1 < nk) {
      stage_tile(Ab + (size_t)(kt + 1) * 32, K, As[cur ^ 1]);
      stage_tile(Wb + (size_t)(kt + 1) * 32, K, Bs[cur ^ 1]);
    }
    const __bf16* la = As[cur] + wm * (64 * 32);
    const __bf16* lb = Bs[cur] + wn * (32 * 32);
    v16bf af[4], bf2[2];
#pragma unroll
    for (int i = 0; i < 4; ++i) af[i] = load_a_frag(la + i * (16 * 32), 32);
#pragma unroll
    for (int j = 0; j < 2; ++j) bf2[j] = load_b_frag(lb + j * (16 * 32), 32);
#pragma unroll
    for (int i = 0; i < 4; ++i)
#pragma unroll
      for (int j = 0; j < 2; ++j)
        acc[i][j] = wmma_bf16(af[i], bf2[j], acc[i][j]);
    __syncthreads();  // done reading buf[cur] before it is refilled at kt+2
  }

  int nin = lane & 15, hh = lane >> 4;
#pragma unroll
  for (int i = 0; i < 4; ++i) {
#pragma unroll
    for (int j = 0; j < 2; ++j) {
#pragma unroll
      for (int r = 0; r < 8; ++r) {
        int m = mbase + i * 16 + r + 8 * hh;
        int n = nbase + j * 16 + nin;
        float v = acc[i][j][r];
        if (EPI == 1) {
          size_t idx = (size_t)m * N + n;
          outf[idx] = resid[idx] + v;
        } else {
          if (EPI == 2) v = gelu_exact(v);
          __bf16 bv = (__bf16)v;
          if (STORE == 0) {
            outb[(size_t)m * N + n] = bv;
          } else {
            int b = m >> 11, s = m & (kS - 1);   // S = 2048
            int hd = n >> 6, d = n & (kDK - 1);  // dk = 64
            if (STORE == 1)
              outb[((size_t)(b * kH + hd) * kS + s) * kDK + d] = bv;
            else
              outb[((size_t)(b * kH + hd) * kDK + d) * kS + s] = bv;
          }
        }
      }
    }
  }
}

// ---------------------------------------------------------------------------
// Flash attention, causal. q,k: [BH,S,64] bf16; vt: [BH,64,S] bf16.
// Block = 8 waves; each wave owns 16 query rows (independent trip counts ->
// wave-level sync only). kk-tiles of 32; P transposed via per-wave LDS tile.
// out: [B,S,H*64] bf16.
// ---------------------------------------------------------------------------
__global__ __launch_bounds__(256)
void attn_kernel(const __bf16* __restrict__ q, const __bf16* __restrict__ k,
                 const __bf16* __restrict__ vt, __bf16* __restrict__ out) {
  int wave = threadIdx.x >> 5;
  int lane = threadIdx.x & 31;
  int tiles_per_bh = kS / 128;  // 16
  int bh = blockIdx.x / tiles_per_bh;
  int q0 = (blockIdx.x % tiles_per_bh) * 128 + wave * 16;
  int b = bh / kH, head = bh % kH;

  const __bf16* qp = q + (size_t)bh * kS * kDK;
  const __bf16* kp = k + (size_t)bh * kS * kDK;
  const __bf16* vp = vt + (size_t)bh * kDK * kS;

  v16bf aq0 = load_a_frag(qp + (size_t)q0 * kDK, kDK);       // d = 0..31
  v16bf aq1 = load_a_frag(qp + (size_t)q0 * kDK + 32, kDK);  // d = 32..63

  v8f acc[4] = {};
  float mrow[8], lrow[8];
#pragma unroll
  for (int r = 0; r < 8; ++r) { mrow[r] = -__builtin_inff(); lrow[r] = 0.0f; }

  __shared__ __align__(64) __bf16 pbuf[8][16 * 32];
  __bf16* pb = pbuf[wave];

  int nin = lane & 15, hh = lane >> 4;
  const float sc = 0.125f;  // 1/sqrt(64)
  int nkt = (q0 + 16 + 31) >> 5;

  for (int kt = 0; kt < nkt; ++kt) {
    int kkb = kt * 32;
    v8f s0 = {}, s1 = {};
    v16bf b00 = load_b_frag(kp + (size_t)kkb * kDK, kDK);
    v16bf b01 = load_b_frag(kp + (size_t)kkb * kDK + 32, kDK);
    v16bf b10 = load_b_frag(kp + (size_t)(kkb + 16) * kDK, kDK);
    v16bf b11 = load_b_frag(kp + (size_t)(kkb + 16) * kDK + 32, kDK);
    s0 = wmma_bf16(aq0, b00, s0);
    s0 = wmma_bf16(aq1, b01, s0);
    s1 = wmma_bf16(aq0, b10, s1);
    s1 = wmma_bf16(aq1, b11, s1);

#pragma unroll
    for (int r = 0; r < 8; ++r) {
      int row = q0 + r + 8 * hh;
      int c0 = kkb + nin, c1 = c0 + 16;
      float x0 = (c0 <= row) ? s0[r] * sc : -__builtin_inff();
      float x1 = (c1 <= row) ? s1[r] * sc : -__builtin_inff();
      float mx = fmaxf(x0, x1);
      mx = fmaxf(mx, __shfl_xor(mx, 1, 32));
      mx = fmaxf(mx, __shfl_xor(mx, 2, 32));
      mx = fmaxf(mx, __shfl_xor(mx, 4, 32));
      mx = fmaxf(mx, __shfl_xor(mx, 8, 32));
      float mnew = fmaxf(mrow[r], mx);
      float rescale = expf(mrow[r] - mnew);
      float p0 = expf(x0 - mnew);
      float p1 = expf(x1 - mnew);
      float ps = p0 + p1;
      ps += __shfl_xor(ps, 1, 32);
      ps += __shfl_xor(ps, 2, 32);
      ps += __shfl_xor(ps, 4, 32);
      ps += __shfl_xor(ps, 8, 32);
      lrow[r] = lrow[r] * rescale + ps;
      mrow[r] = mnew;
#pragma unroll
      for (int j = 0; j < 4; ++j) acc[j][r] *= rescale;
      pb[(r + 8 * hh) * 32 + nin] = (__bf16)p0;
      pb[(r + 8 * hh) * 32 + 16 + nin] = (__bf16)p1;
    }

    __builtin_amdgcn_wave_barrier();
    asm volatile("s_wait_dscnt 0" ::: "memory");

    v16bf pa = load_a_frag(pb, 32);
#pragma unroll
    for (int j = 0; j < 4; ++j) {
      v16bf bv = load_b_frag(vp + (size_t)(j * 16) * kS + kkb, kS);
      acc[j] = wmma_bf16(pa, bv, acc[j]);
    }
    __builtin_amdgcn_wave_barrier();
  }

#pragma unroll
  for (int j = 0; j < 4; ++j) {
#pragma unroll
    for (int r = 0; r < 8; ++r) {
      int srow = q0 + r + 8 * hh;
      int d = j * 16 + nin;
      float o = acc[j][r] / lrow[r];
      out[((size_t)(b * kS + srow)) * kD + head * kDK + d] = (__bf16)o;
    }
  }
}

// ---------------------------------------------------------------------------
// Host orchestration
// ---------------------------------------------------------------------------
extern "C" void kernel_launch(void* const* d_in, const int* in_sizes, int n_in,
                              void* d_out, int out_size, void* d_ws, size_t ws_size,
                              hipStream_t stream) {
  (void)in_sizes; (void)n_in; (void)out_size; (void)ws_size;
  const float* x  = (const float*)d_in[0];
  const float* wq = (const float*)d_in[1];
  const float* wk = (const float*)d_in[2];
  const float* wv = (const float*)d_in[3];
  const float* wo = (const float*)d_in[4];
  const float* w1 = (const float*)d_in[5];
  const float* w2 = (const float*)d_in[6];
  const float* g1 = (const float*)d_in[7];
  const float* g2 = (const float*)d_in[8];

  char* w = (char*)d_ws;
  size_t off = 0;
  auto take = [&](size_t bytes) -> char* {
    char* p = w + off;
    off += (bytes + 255) & ~(size_t)255;
    return p;
  };
  __bf16* wq_b = (__bf16*)take((size_t)kD * kD * 2);
  __bf16* wk_b = (__bf16*)take((size_t)kD * kD * 2);
  __bf16* wv_b = (__bf16*)take((size_t)kD * kD * 2);
  __bf16* wo_b = (__bf16*)take((size_t)kD * kD * 2);
  __bf16* w1_b = (__bf16*)take((size_t)kF * kD * 2);
  __bf16* w2_b = (__bf16*)take((size_t)kD * kF * 2);
  __bf16* h_b  = (__bf16*)take((size_t)kM * kD * 2);   // rmsnorm1(x)
  __bf16* q_b  = (__bf16*)take((size_t)kBH * kS * kDK * 2);
  __bf16* k_b  = (__bf16*)take((size_t)kBH * kS * kDK * 2);
  __bf16* vt_b = (__bf16*)take((size_t)kBH * kDK * kS * 2);
  __bf16* a_b  = (__bf16*)take((size_t)kM * kD * 2);   // attention output
  float*  x1   = (float*) take((size_t)kM * kD * 4);   // x + attn
  __bf16* h2_b = (__bf16*)take((size_t)kM * kD * 2);   // rmsnorm2(x1)
  __bf16* ff_b = (__bf16*)take((size_t)kM * kF * 2);   // gelu(h2 @ w1^T)

  dim3 blk(256);

  // --- weight conversion (deterministic, every call) ---
  cvt_bf16_kernel<<<dim3(1024), blk, 0, stream>>>(wq, wq_b, kD * kD);
  cvt_bf16_kernel<<<dim3(1024), blk, 0, stream>>>(wk, wk_b, kD * kD);
  cvt_bf16_kernel<<<dim3(1024), blk, 0, stream>>>(wv, wv_b, kD * kD);
  cvt_bf16_kernel<<<dim3(1024), blk, 0, stream>>>(wo, wo_b, kD * kD);
  cvt_bf16_kernel<<<dim3(2048), blk, 0, stream>>>(w1, w1_b, kF * kD);
  cvt_bf16_kernel<<<dim3(2048), blk, 0, stream>>>(w2, w2_b, kD * kF);

  // --- attention sublayer ---
  rmsnorm_bf16_kernel<<<dim3(kM), blk, 0, stream>>>(x, g1, h_b);

  dim3 gD(kD / 128, kM / 128);   // (8, 32)
  dim3 gF(kF / 128, kM / 128);   // (32, 32)
  gemm_bf16_kernel<1, 0><<<gD, blk, 0, stream>>>(h_b, wq_b, kM, kD, kD, nullptr, q_b, nullptr);
  gemm_bf16_kernel<1, 0><<<gD, blk, 0, stream>>>(h_b, wk_b, kM, kD, kD, nullptr, k_b, nullptr);
  gemm_bf16_kernel<2, 0><<<gD, blk, 0, stream>>>(h_b, wv_b, kM, kD, kD, nullptr, vt_b, nullptr);

  attn_kernel<<<dim3(kBH * (kS / 128)), blk, 0, stream>>>(q_b, k_b, vt_b, a_b);

  // x1 = x + attn_out @ wo^T
  gemm_bf16_kernel<0, 1><<<gD, blk, 0, stream>>>(a_b, wo_b, kM, kD, kD, x1, nullptr, x);

  // --- FFN sublayer ---
  rmsnorm_bf16_kernel<<<dim3(kM), blk, 0, stream>>>(x1, g2, h2_b);
  gemm_bf16_kernel<0, 2><<<gF, blk, 0, stream>>>(h2_b, w1_b, kM, kF, kD, nullptr, ff_b, nullptr);
  gemm_bf16_kernel<0, 1><<<gD, blk, 0, stream>>>(ff_b, w2_b, kM, kD, kF, (float*)d_out, nullptr, x1);
}
